// MoEBlock_16776142258567
// MI455X (gfx1250) — matrix operand order repaired and checked by reading, not compiled, not run on hardware
//
#include <hip/hip_runtime.h>
#include <math.h>

// ---------------- problem constants (from reference) ----------------
#define T_TOK 2048      // B*S
#define DIM   1024      // D
#define FF    4096      // F
#define NEXP  8         // E
#define OUT_ELEMS (T_TOK * DIM)          // 2,097,152 floats of `out`
// router_logits follow at d_out + OUT_ELEMS (T_TOK * NEXP floats)

// GEMM tiling: 64 tokens x 128 N per block, K panels of 64, 8 waves/block
#define BM 64
#define BN 128
#define BK 64

typedef __attribute__((ext_vector_type(16))) __bf16 v16bf;
typedef __attribute__((ext_vector_type(8)))  __bf16 v8bf;
typedef __attribute__((ext_vector_type(8)))  float  v8f;
typedef int moe_v4i __attribute__((vector_size(16)));   // int4 vector for async builtin

// ---------------- gfx1250 async global->LDS copy (guarded) ----------------
#if __has_builtin(__builtin_amdgcn_global_load_async_to_lds_b128)
#define MOE_ASYNC_LDS 1
template <int OFF>
__device__ inline void async_b128(const __bf16* g, __bf16* lds) {
    // copies 16 bytes MEM[g+OFF] -> LDS[lds+OFF], tracked on ASYNCcnt
    __builtin_amdgcn_global_load_async_to_lds_b128(
        (__attribute__((address_space(1))) moe_v4i*)g,
        (__attribute__((address_space(3))) moe_v4i*)lds, OFF, 0);
}
#else
#define MOE_ASYNC_LDS 0
#endif

__device__ inline void wait_async0() {
#if __has_builtin(__builtin_amdgcn_s_wait_asynccnt)
    __builtin_amdgcn_s_wait_asynccnt(0);
#elif MOE_ASYNC_LDS
    asm volatile("s_wait_asynccnt 0" ::: "memory");
#endif
}

// ---------------- small utility kernels ----------------
__global__ void moe_zero_init(float* __restrict__ out, int n, int* __restrict__ counts) {
    int i = blockIdx.x * blockDim.x + threadIdx.x;
    if (i < n) out[i] = 0.0f;
    if (i < NEXP) counts[i] = 0;
}

__global__ void moe_cvt_x(const float* __restrict__ x, __bf16* __restrict__ xb, int n) {
    int i = blockIdx.x * blockDim.x + threadIdx.x;
    if (i < n) xb[i] = (__bf16)x[i];
}

__device__ inline float wave_reduce_add(float v) {
    #pragma unroll
    for (int off = 16; off > 0; off >>= 1) v += __shfl_xor(v, off, 32);
    return v;
}

// pack two fp32 into one dword of 2x bf16 (lowers to v_cvt_pk_bf16_f32)
__device__ inline unsigned pack_bf16x2(float lo, float hi) {
    union { __bf16 h[2]; unsigned u; } p;
    p.h[0] = (__bf16)lo;
    p.h[1] = (__bf16)hi;
    return p.u;
}

// ---------------- router: logits, softmax, top-2, expert lists ----------------
__global__ __launch_bounds__(256) void moe_router(
    const float* __restrict__ x,        // [T, D]
    const float* __restrict__ gate_w,   // [E, D]
    float* __restrict__ logits_out,     // [T, E]
    int* __restrict__ counts,           // [E]
    int2* __restrict__ lists)           // [E][T]  (.x = token*2+k, .y = bits(weight))
{
    const int lane = threadIdx.x & 31;
    const int wave = threadIdx.x >> 5;
    const int t = blockIdx.x * 8 + wave;
    if (t >= T_TOK) return;

    float xv[DIM / 32];
    const float* xr = x + (size_t)t * DIM;
    #pragma unroll
    for (int j = 0; j < DIM / 32; ++j) xv[j] = xr[j * 32 + lane];

    float lg[NEXP];
    #pragma unroll
    for (int e = 0; e < NEXP; ++e) {
        const float* gw = gate_w + (size_t)e * DIM;
        float s = 0.0f;
        #pragma unroll
        for (int j = 0; j < DIM / 32; ++j) s = fmaf(xv[j], gw[j * 32 + lane], s);
        lg[e] = wave_reduce_add(s);
    }

    if (lane == 0) {
        float* lo = logits_out + (size_t)t * NEXP;
        #pragma unroll
        for (int e = 0; e < NEXP; ++e) lo[e] = lg[e];

        int i0 = 0; float l0 = lg[0];
        #pragma unroll
        for (int e = 1; e < NEXP; ++e) if (lg[e] > l0) { l0 = lg[e]; i0 = e; }
        int i1 = (i0 == 0) ? 1 : 0; float l1 = lg[i1];
        #pragma unroll
        for (int e = 0; e < NEXP; ++e)
            if (e != i0 && lg[e] > l1) { l1 = lg[e]; i1 = e; }

        float w0 = 1.0f / (1.0f + __expf(l1 - l0));
        float w1 = 1.0f - w0;

        int s0 = atomicAdd(&counts[i0], 1);
        lists[i0 * T_TOK + s0] = make_int2(t * 2 + 0, __float_as_int(w0));
        int s1 = atomicAdd(&counts[i1], 1);
        lists[i1 * T_TOK + s1] = make_int2(t * 2 + 1, __float_as_int(w1));
    }
}

// ---------------- shared GEMM staging ----------------
// B panel: fp32 [BK x BN] -> bf16 Bs[n][k]; k-pair packed dword stores.
// Bg_kn points at element (k0, n0).
__device__ inline void stage_B(const float* __restrict__ Bg_kn, int ldb,
                               __bf16 (*Bs)[BK], int tid) {
    #pragma unroll
    for (int i = 0; i < 4; ++i) {
        int g = tid * 4 + i;                 // 1024 groups: (kpair, nquad)
        int n = (g & 31) * 4;
        int k = (g >> 5) * 2;
        const float* r0 = Bg_kn + (size_t)k * ldb + n;
        float4 a = *(const float4*)r0;
        float4 b = *(const float4*)(r0 + ldb);
        *(unsigned*)&Bs[n + 0][k] = pack_bf16x2(a.x, b.x);
        *(unsigned*)&Bs[n + 1][k] = pack_bf16x2(a.y, b.y);
        *(unsigned*)&Bs[n + 2][k] = pack_bf16x2(a.z, b.z);
        *(unsigned*)&Bs[n + 3][k] = pack_bf16x2(a.w, b.w);
    }
}

// A panel: 32 bytes per thread from the (loop-invariant) gathered row.
__device__ inline void stage_A(const __bf16* __restrict__ aSrc /* row base + ac */,
                               int k0, __bf16* aDst /* &As[ar][ac] */) {
#if MOE_ASYNC_LDS
    async_b128<0>(aSrc + k0, aDst);          // global_load_async_to_lds_b128
    async_b128<16>(aSrc + k0, aDst);
#else
    *(uint4*)aDst       = *(const uint4*)(aSrc + k0);
    *(uint4*)(aDst + 8) = *(const uint4*)(aSrc + k0 + 8);
#endif
}

// ---------------- GEMM1: h = gelu(x @ fc_w + fc_b) -> bf16 ----------------
__global__ __launch_bounds__(256) void moe_fc_gelu(
    const float* __restrict__ fc_w,     // [E, D, F]
    const float* __restrict__ fc_b,     // [E, F]
    const __bf16* __restrict__ xb,      // [T, D]
    const int* __restrict__ counts,
    const int2* __restrict__ lists,
    __bf16* __restrict__ hb)            // [2T, F], row = token*2+k
{
    __shared__ __bf16 As[BM][BK];       // 8 KB
    __shared__ __bf16 Bs[BN][BK];       // 16 KB
    __shared__ int rowA[BM];

    const int e = blockIdx.z;
    const int cnt = counts[e];
    const int mBase = blockIdx.y * BM;
    if (mBase >= cnt) return;
    const int n0 = blockIdx.x * BN;
    const int tid  = threadIdx.x;
    const int lane = tid & 31;
    const int wave = tid >> 5;

    if (tid < BM) {
        int idx = mBase + tid;
        rowA[tid] = lists[e * T_TOK + (idx < cnt ? idx : mBase)].x;
    }
    __syncthreads();

    // loop-invariant A gather pointer for this thread
    const int ar = tid >> 2;
    const int ac = (tid & 3) * 16;
    const __bf16* aSrc = xb + (size_t)(rowA[ar] >> 1) * DIM + ac;
    __bf16* aDst = &As[ar][ac];

    const float* Bg = fc_w + (size_t)e * DIM * FF + n0;
    v8f acc[4] = {v8f{}, v8f{}, v8f{}, v8f{}};

    const int kh = lane >> 4;
    const int ml = lane & 15;

    for (int k0 = 0; k0 < DIM; k0 += BK) {
        stage_A(aSrc, k0, aDst);
        stage_B(Bg + (size_t)k0 * FF, FF, Bs, tid);
        wait_async0();
        __syncthreads();

        if (k0 + BK < DIM)  // global_prefetch_b8 for next weight panel
            __builtin_prefetch(Bg + (size_t)(k0 + BK + (tid >> 2)) * FF, 0, 1);

        #pragma unroll
        for (int p = 0; p < 2; ++p) {   // two 32-K phases per panel
            v16bf bf = *(const v16bf*)&Bs[wave * 16 + ml][p * 32 + kh * 16];
            #pragma unroll
            for (int mi = 0; mi < 4; ++mi) {
                union { v16bf v; v8bf h[2]; } a;
                a.h[0] = *(const v8bf*)&As[mi * 16 + ml][p * 32 + kh * 16];
                a.h[1] = *(const v8bf*)&As[mi * 16 + ml][p * 32 + kh * 16 + 8];
                acc[mi] = __builtin_amdgcn_wmma_f32_16x16x32_bf16(
                    false, a.v, false, bf, (short)0, acc[mi], false, false);
            }
        }
        __syncthreads();
    }

    // C layout: lanes 0-15 -> M=r, lanes 16-31 -> M=r+8; N = lane&15
    const int mAdd = (lane >= 16) ? 8 : 0;
    const int nG = n0 + wave * 16 + ml;
    const float bias = fc_b[e * FF + nG];
    #pragma unroll
    for (int mi = 0; mi < 4; ++mi) {
        #pragma unroll
        for (int r = 0; r < 8; ++r) {
            int m = mi * 16 + r + mAdd;
            if (mBase + m < cnt) {
                float v = acc[mi][r] + bias;
                v = 0.5f * v * (1.0f + erff(v * 0.70710678118f));   // exact GELU
                hb[(size_t)rowA[m] * FF + nG] = (__bf16)v;
            }
        }
    }
}

// ---------------- GEMM2: out += w * (h @ proj_w + proj_b) ----------------
__global__ __launch_bounds__(256) void moe_proj_scatter(
    const float* __restrict__ proj_w,   // [E, F, D]
    const float* __restrict__ proj_b,   // [E, D]
    const __bf16* __restrict__ hb,      // [2T, F]
    const int* __restrict__ counts,
    const int2* __restrict__ lists,
    float* __restrict__ out)            // [T, D]
{
    __shared__ __bf16 As[BM][BK];
    __shared__ __bf16 Bs[BN][BK];
    __shared__ int rowA[BM];
    __shared__ float rowW[BM];

    const int e = blockIdx.z;
    const int cnt = counts[e];
    const int mBase = blockIdx.y * BM;
    if (mBase >= cnt) return;
    const int n0 = blockIdx.x * BN;
    const int tid  = threadIdx.x;
    const int lane = tid & 31;
    const int wave = tid >> 5;

    if (tid < BM) {
        int idx = mBase + tid;
        int2 ent = lists[e * T_TOK + (idx < cnt ? idx : mBase)];
        rowA[tid] = ent.x;
        rowW[tid] = __int_as_float(ent.y);
    }
    __syncthreads();

    const int ar = tid >> 2;
    const int ac = (tid & 3) * 16;
    const __bf16* aSrc = hb + (size_t)rowA[ar] * FF + ac;
    __bf16* aDst = &As[ar][ac];

    const float* Bg = proj_w + (size_t)e * FF * DIM + n0;
    v8f acc[4] = {v8f{}, v8f{}, v8f{}, v8f{}};

    const int kh = lane >> 4;
    const int ml = lane & 15;

    for (int k0 = 0; k0 < FF; k0 += BK) {
        stage_A(aSrc, k0, aDst);
        stage_B(Bg + (size_t)k0 * DIM, DIM, Bs, tid);
        wait_async0();
        __syncthreads();

        if (k0 + BK < FF)
            __builtin_prefetch(Bg + (size_t)(k0 + BK + (tid >> 2)) * DIM, 0, 1);

        #pragma unroll
        for (int p = 0; p < 2; ++p) {
            v16bf bf = *(const v16bf*)&Bs[wave * 16 + ml][p * 32 + kh * 16];
            #pragma unroll
            for (int mi = 0; mi < 4; ++mi) {
                union { v16bf v; v8bf h[2]; } a;
                a.h[0] = *(const v8bf*)&As[mi * 16 + ml][p * 32 + kh * 16];
                a.h[1] = *(const v8bf*)&As[mi * 16 + ml][p * 32 + kh * 16 + 8];
                acc[mi] = __builtin_amdgcn_wmma_f32_16x16x32_bf16(
                    false, a.v, false, bf, (short)0, acc[mi], false, false);
            }
        }
        __syncthreads();
    }

    const int mAdd = (lane >= 16) ? 8 : 0;
    const int nG = n0 + wave * 16 + ml;
    const float pb = proj_b[e * DIM + nG];
    #pragma unroll
    for (int mi = 0; mi < 4; ++mi) {
        #pragma unroll
        for (int r = 0; r < 8; ++r) {
            int m = mi * 16 + r + mAdd;
            if (mBase + m < cnt) {
                int tok = rowA[m] >> 1;
                atomicAdd(out + (size_t)tok * DIM + nG, rowW[m] * (acc[mi][r] + pb));
            }
        }
    }
}

// ---------------- launcher ----------------
extern "C" void kernel_launch(void* const* d_in, const int* in_sizes, int n_in,
                              void* d_out, int out_size, void* d_ws, size_t ws_size,
                              hipStream_t stream) {
    const float* x      = (const float*)d_in[0];   // [2,1024,1024]
    const float* gate_w = (const float*)d_in[1];   // [8,1024]
    const float* fc_w   = (const float*)d_in[2];   // [8,1024,4096]
    const float* fc_b   = (const float*)d_in[3];   // [8,4096]
    const float* proj_w = (const float*)d_in[4];   // [8,4096,1024]
    const float* proj_b = (const float*)d_in[5];   // [8,1024]

    float* out    = (float*)d_out;                 // [T, D]
    float* logits = out + OUT_ELEMS;               // [T, E]

    char* ws = (char*)d_ws;
    __bf16* xb   = (__bf16*)ws;                                   // 4 MB
    size_t off   = (size_t)T_TOK * DIM * sizeof(__bf16);
    int*   counts = (int*)(ws + off);
    int2*  lists  = (int2*)(ws + off + 256);                      // 128 KB
    __bf16* hb    = (__bf16*)(ws + off + 256 + (size_t)NEXP * T_TOK * sizeof(int2)); // 32 MB

    moe_zero_init<<<(OUT_ELEMS + 255) / 256, 256, 0, stream>>>(out, OUT_ELEMS, counts);
    moe_cvt_x<<<(T_TOK * DIM + 255) / 256, 256, 0, stream>>>(x, xb, T_TOK * DIM);
    moe_router<<<T_TOK / 8, 256, 0, stream>>>(x, gate_w, logits, counts, lists);
    moe_fc_gelu<<<dim3(FF / BN, T_TOK / BM, NEXP), 256, 0, stream>>>(
        fc_w, fc_b, xb, counts, lists, hb);
    moe_proj_scatter<<<dim3(DIM / BN, T_TOK / BM, NEXP), 256, 0, stream>>>(
        proj_w, proj_b, hb, counts, lists, out);
}
